// HNet_3367254360095
// MI455X (gfx1250) — matrix operand order repaired and checked by reading, not compiled, not run on hardware
//
#include <hip/hip_runtime.h>

#define NBATCH 4
#define SEQ    1024
#define DMODEL 512
#define NHEAD  4
#define HDIM   128
#define DFFN   2048
#define NEXP   4
#define NTOK   (NBATCH * SEQ)   // 4096

typedef __attribute__((ext_vector_type(16))) __bf16 v16bf;
typedef __attribute__((ext_vector_type(8)))  __bf16 v8bf;
typedef __attribute__((ext_vector_type(4)))  __bf16 v4bf;
typedef __attribute__((ext_vector_type(8)))  float  v8f;

union U16 { v16bf v; v8bf h[2]; };

// ---------------------------------------------------------------------------
// f32 -> bf16 (weights & activations feeding WMMA)
// ---------------------------------------------------------------------------
__global__ void k_f2bf(const float* __restrict__ s, __bf16* __restrict__ d, int n) {
  int i = (blockIdx.x * blockDim.x + threadIdx.x) * 4;
  if (i >= n) return;
  v4bf o;
  o[0] = (__bf16)s[i + 0]; o[1] = (__bf16)s[i + 1];
  o[2] = (__bf16)s[i + 2]; o[3] = (__bf16)s[i + 3];
  *(v4bf*)(d + i) = o;
}

// ---------------------------------------------------------------------------
// WMMA GEMM:  C[n,m] = sum_k A[n,k] * B[m,k]   (A: NxK bf16, B: MxK bf16)
// Block tile: 128 tokens x 256 cols x 32 K. 256 threads = 8 wave32 waves in a
// 2x4 wave grid; each wave owns a 64x64 output tile -> 16 v8f accumulators
// (128 VGPRs) and issues 16 v_wmma_f32_16x16x32_bf16 per K step against
// 16 ds_load_b128 fragment reads (4 A frags + 4 B frags, each reused 4x).
// LDS tiles (A 8KB + B 16KB, double buffered = 48KB) are filled with
// global_load_async_to_lds_b128 (6 per thread per tile), pipelined on
// ASYNCcnt with per-thread global byte offsets advanced by +64B per step.
// EPI: 0 = store, 1 = store + residual, 2 = C += scale[n]*acc (scale null->1)
// ---------------------------------------------------------------------------
template <int EPI>
__global__ __launch_bounds__(256) void k_gemm(
    const __bf16* __restrict__ A, const __bf16* __restrict__ Bw,
    float* __restrict__ C, const float* __restrict__ R,
    const float* __restrict__ scale, int sstride, int N, int M, int K) {
  // buf layout (bytes): A tile [0, 8192), B tile [8192, 24576); x2 buffers
  __shared__ __align__(16) __bf16 smem[24576];

  const int t    = threadIdx.x;
  const int n0   = blockIdx.x * 128;
  const int m0   = blockIdx.y * 256;
  const int wid  = t >> 5, lane = t & 31;
  const int wm   = wid >> 2;   // 0..1 along tokens (64 rows each)
  const int wn   = wid & 3;    // 0..3 along cols   (64 cols each)
  const int r    = lane & 15, hc = lane >> 4;

  // per-thread async-copy source offsets (bytes), advanced +64 per K step
  unsigned gA[2], gB[4];
  unsigned lA[2], lB[4];
#pragma unroll
  for (int i = 0; i < 2; ++i) {             // A: 512 chunks of 16B, 2/thread
    const int c = 2 * t + i;
    gA[i] = (unsigned)(((n0 + (c >> 2)) * K + (c & 3) * 8) * 2);
    lA[i] = (unsigned)(c * 16);
  }
#pragma unroll
  for (int i = 0; i < 4; ++i) {             // B: 1024 chunks of 16B, 4/thread
    const int c = 4 * t + i;
    gB[i] = (unsigned)(((m0 + (c >> 2)) * K + (c & 3) * 8) * 2);
    lB[i] = (unsigned)(8192 + c * 16);
  }

  auto prefetch = [&](int buf) {
    const unsigned bb = (unsigned)(buf * 24576);
#pragma unroll
    for (int i = 0; i < 2; ++i) {
      unsigned l = bb + lA[i];
      asm volatile("global_load_async_to_lds_b128 %0, %1, %2"
                   :: "v"(l), "v"(gA[i]), "s"(A) : "memory");
      gA[i] += 64;
    }
#pragma unroll
    for (int i = 0; i < 4; ++i) {
      unsigned l = bb + lB[i];
      asm volatile("global_load_async_to_lds_b128 %0, %1, %2"
                   :: "v"(l), "v"(gB[i]), "s"(Bw) : "memory");
      gB[i] += 64;
    }
  };

  v8f acc[4][4] = {};
  const int KT = K >> 5;
  prefetch(0);
  for (int kt = 0; kt < KT; ++kt) {
    const int buf = kt & 1;
    if (kt + 1 < KT) {
      prefetch(buf ^ 1);
      asm volatile("s_wait_asynccnt 6" ::: "memory");  // current tile's 6 done
    } else {
      asm volatile("s_wait_asynccnt 0" ::: "memory");
    }
    __syncthreads();

    v16bf af[4], bfm[4];
#pragma unroll
    for (int i = 0; i < 4; ++i) {  // A 16x32 frag: lane r row; chunks 8hc, 16+8hc
      const __bf16* p = smem + buf * 12288 + (wm * 64 + i * 16 + r) * 32;
      U16 u;
      u.h[0] = *(const v8bf*)(p + 8 * hc);
      u.h[1] = *(const v8bf*)(p + 16 + 8 * hc);
      af[i] = u.v;
    }
#pragma unroll
    for (int j = 0; j < 4; ++j) {  // B 32x16 frag: lane r col; chunks 8hc, 16+8hc
      const __bf16* p = smem + buf * 12288 + 4096 + (wn * 64 + j * 16 + r) * 32;
      U16 u;
      u.h[0] = *(const v8bf*)(p + 8 * hc);
      u.h[1] = *(const v8bf*)(p + 16 + 8 * hc);
      bfm[j] = u.v;
    }
#pragma unroll
    for (int i = 0; i < 4; ++i)
#pragma unroll
      for (int j = 0; j < 4; ++j)
        acc[i][j] = __builtin_amdgcn_wmma_f32_16x16x32_bf16(
            false, af[i], false, bfm[j], (short)0, acc[i][j], false, false);
    __syncthreads();
  }

#pragma unroll
  for (int i = 0; i < 4; ++i)
#pragma unroll
    for (int j = 0; j < 4; ++j)
#pragma unroll
      for (int v = 0; v < 8; ++v) {
        const int n = n0 + wm * 64 + i * 16 + v + 8 * hc;  // C layout: VGPR v -> M row
        const int m = m0 + wn * 64 + j * 16 + r;           // lane -> N col
        const size_t o = (size_t)n * M + m;
        float val = acc[i][j][v];
        if (EPI == 0) {
          C[o] = val;
        } else if (EPI == 1) {
          C[o] = val + R[o];
        } else {
          float sc = scale ? scale[(size_t)n * sstride] : 1.0f;
          C[o] += sc * val;
        }
      }
}

// ---------------------------------------------------------------------------
// RMSNorm: one row (D=512) per block of 128 threads. Writes f32 and bf16.
// ---------------------------------------------------------------------------
__global__ __launch_bounds__(128) void k_rmsnorm(const float* __restrict__ X,
                                                 float* __restrict__ Of,
                                                 __bf16* __restrict__ Ob) {
  __shared__ float red[4];
  const int row = blockIdx.x, t = threadIdx.x;
  const float* x = X + (size_t)row * DMODEL + 4 * t;
  float a0 = x[0], a1 = x[1], a2 = x[2], a3 = x[3];
  float ss = a0 * a0 + a1 * a1 + a2 * a2 + a3 * a3;
#pragma unroll
  for (int o = 16; o > 0; o >>= 1) ss += __shfl_xor(ss, o, 32);
  if ((t & 31) == 0) red[t >> 5] = ss;
  __syncthreads();
  float inv = rsqrtf((red[0] + red[1] + red[2] + red[3]) * (1.0f / DMODEL) + 1e-6f);
  float y0 = a0 * inv, y1 = a1 * inv, y2 = a2 * inv, y3 = a3 * inv;
  float4 yf = {y0, y1, y2, y3};
  *(float4*)(Of + (size_t)row * DMODEL + 4 * t) = yf;
  v4bf yb; yb[0] = (__bf16)y0; yb[1] = (__bf16)y1; yb[2] = (__bf16)y2; yb[3] = (__bf16)y3;
  *(v4bf*)(Ob + (size_t)row * DMODEL + 4 * t) = yb;
}

// ---------------------------------------------------------------------------
// RoPE(q) + 4x4 head-attention per position. One wave per (b,s).
// Output written in the reference's (b,h,s,hd) flat layout, as bf16.
// ---------------------------------------------------------------------------
__global__ __launch_bounds__(128) void k_attn(const float* __restrict__ Q,
                                              const float* __restrict__ KV,
                                              __bf16* __restrict__ O) {
  const int pos = blockIdx.x * 4 + (threadIdx.x >> 5);
  const int lane = threadIdx.x & 31;
  const int b = pos >> 10, s = pos & (SEQ - 1);
  const float* q = Q + (size_t)pos * DMODEL;
  const float* kv = KV + (size_t)pos * (2 * DMODEL);

  float qr[4][4], kr[4][4], vr[4][4];
#pragma unroll
  for (int h = 0; h < 4; ++h) {
#pragma unroll
    for (int u = 0; u < 4; ++u) {
      qr[h][u] = q[h * HDIM + 4 * lane + u];
      kr[h][u] = kv[h * 2 * HDIM + 4 * lane + u];
      vr[h][u] = kv[h * 2 * HDIM + HDIM + 4 * lane + u];
    }
#pragma unroll
    for (int u = 0; u < 4; u += 2) {  // rope pairs (even,odd); q only
      float j = (float)(2 * lane + (u >> 1));
      float ang = (float)s * __powf(10000.0f, -j * (1.0f / 64.0f));
      float sn, cs; __sincosf(ang, &sn, &cs);
      float x0 = qr[h][u], x1 = qr[h][u + 1];
      qr[h][u] = x0 * cs - x1 * sn;
      qr[h][u + 1] = x0 * sn + x1 * cs;
    }
  }
  float att[4][4];
#pragma unroll
  for (int i = 0; i < 4; ++i)
#pragma unroll
    for (int j = 0; j < 4; ++j) {
      float p = qr[i][0] * kr[j][0] + qr[i][1] * kr[j][1] +
                qr[i][2] * kr[j][2] + qr[i][3] * kr[j][3];
#pragma unroll
      for (int o = 16; o > 0; o >>= 1) p += __shfl_xor(p, o, 32);
      att[i][j] = p * 0.08838834764831845f;  // 1/sqrt(128)
    }
#pragma unroll
  for (int i = 0; i < 4; ++i) {
    float mx = fmaxf(fmaxf(att[i][0], att[i][1]), fmaxf(att[i][2], att[i][3]));
    float sum = 0.f;
#pragma unroll
    for (int j = 0; j < 4; ++j) { att[i][j] = __expf(att[i][j] - mx); sum += att[i][j]; }
    float inv = 1.0f / sum;
#pragma unroll
    for (int j = 0; j < 4; ++j) att[i][j] *= inv;
  }
#pragma unroll
  for (int i = 0; i < 4; ++i) {
    v4bf o4;
#pragma unroll
    for (int u = 0; u < 4; ++u) {
      float a = att[i][0] * vr[0][u] + att[i][1] * vr[1][u] +
                att[i][2] * vr[2][u] + att[i][3] * vr[3][u];
      o4[u] = (__bf16)a;
    }
    *(v4bf*)(O + ((size_t)(b * NHEAD + i) * SEQ + s) * HDIM + 4 * lane) = o4;
  }
}

// ---------------------------------------------------------------------------
// Gating: softmax over 4 logits, top-2, normalized weights. One wave/token.
// ---------------------------------------------------------------------------
__global__ __launch_bounds__(256) void k_gate(const float* __restrict__ XN,
                                              const float* __restrict__ G,
                                              float* __restrict__ WE) {
  const int tok = blockIdx.x * 8 + (threadIdx.x >> 5);
  const int lane = threadIdx.x & 31;
  const float* x = XN + (size_t)tok * DMODEL;
  float lg[NEXP];
#pragma unroll
  for (int e = 0; e < NEXP; ++e) {
    const float* g = G + e * DMODEL;
    float p = 0.f;
    for (int d = lane; d < DMODEL; d += 32) p += x[d] * g[d];
#pragma unroll
    for (int o = 16; o > 0; o >>= 1) p += __shfl_xor(p, o, 32);
    lg[e] = p;
  }
  float mx = fmaxf(fmaxf(lg[0], lg[1]), fmaxf(lg[2], lg[3]));
  float pe[NEXP], sum = 0.f;
#pragma unroll
  for (int e = 0; e < NEXP; ++e) { pe[e] = __expf(lg[e] - mx); sum += pe[e]; }
#pragma unroll
  for (int e = 0; e < NEXP; ++e) pe[e] /= sum;
  int i0 = 0;
#pragma unroll
  for (int e = 1; e < NEXP; ++e) if (pe[e] > pe[i0]) i0 = e;
  int i1 = (i0 == 0) ? 1 : 0;
#pragma unroll
  for (int e = 0; e < NEXP; ++e) if (e != i0 && pe[e] > pe[i1]) i1 = e;
  float wsum = pe[i0] + pe[i1];
  if (lane == 0) {
#pragma unroll
    for (int e = 0; e < NEXP; ++e)
      WE[(size_t)tok * NEXP + e] =
          (e == i0) ? pe[i0] / wsum : ((e == i1) ? pe[i1] / wsum : 0.f);
  }
}

// ---------------------------------------------------------------------------
// gelu(a)*b -> bf16
// ---------------------------------------------------------------------------
__global__ void k_gelu_mul(const float* __restrict__ a, const float* __restrict__ b,
                           __bf16* __restrict__ o, int n) {
  int i = (blockIdx.x * blockDim.x + threadIdx.x) * 4;
  if (i >= n) return;
  v4bf v;
#pragma unroll
  for (int u = 0; u < 4; ++u) {
    float x = a[i + u];
    float g = 0.5f * x * (1.0f + erff(x * 0.70710678118f));
    v[u] = (__bf16)(g * b[i + u]);
  }
  *(v4bf*)(o + i) = v;
}

// ---------------------------------------------------------------------------
// Router: cosine similarity of adjacent projected states. One wave/position.
// ---------------------------------------------------------------------------
__global__ __launch_bounds__(256) void k_router(const float* __restrict__ Qr,
                                                const float* __restrict__ Kr,
                                                float* __restrict__ BP,
                                                int* __restrict__ MK) {
  const int pos = blockIdx.x * 8 + (threadIdx.x >> 5);
  const int lane = threadIdx.x & 31;
  const int s = pos & (SEQ - 1);
  if (s == 0) {
    if (lane == 0) { BP[pos] = 1.0f; MK[pos] = 1; }
    return;
  }
  const float* q = Qr + (size_t)(pos - 1) * DMODEL;
  const float* k = Kr + (size_t)pos * DMODEL;
  float qk = 0.f, qq = 0.f, kk = 0.f;
  for (int d = lane; d < DMODEL; d += 32) {
    float a = q[d], c = k[d];
    qk += a * c; qq += a * a; kk += c * c;
  }
#pragma unroll
  for (int o = 16; o > 0; o >>= 1) {
    qk += __shfl_xor(qk, o, 32);
    qq += __shfl_xor(qq, o, 32);
    kk += __shfl_xor(kk, o, 32);
  }
  float cosv = qk * rsqrtf(fmaxf(qq, 1e-24f)) * rsqrtf(fmaxf(kk, 1e-24f));
  float bp = fminf(fmaxf((1.0f - cosv) * 0.5f, 0.0f), 1.0f);
  if (lane == 0) { BP[pos] = bp; MK[pos] = (bp > 0.5f) ? 1 : 0; }
}

// ---------------------------------------------------------------------------
// Per-batch sequential scan of mask -> cum counts + compacted indices
// ---------------------------------------------------------------------------
__global__ void k_scan_mask(const int* __restrict__ MK, int* __restrict__ CUM,
                            int* __restrict__ IDX, int* __restrict__ CNT) {
  int b = threadIdx.x;
  if (b >= NBATCH) return;
  int c = 0;
  for (int s = 0; s < SEQ; ++s) {
    int m = MK[b * SEQ + s];
    if (m) { IDX[b * SEQ + c] = s; ++c; }
    CUM[b * SEQ + s] = c;
  }
  CNT[b] = c;
}

// ---------------------------------------------------------------------------
// Chunk gather (zero-padded to full length)
// ---------------------------------------------------------------------------
__global__ __launch_bounds__(128) void k_chunk(const float* __restrict__ ENC,
                                               const int* __restrict__ IDX,
                                               const int* __restrict__ CNT,
                                               float* __restrict__ COMP) {
  const int rowi = blockIdx.x;                 // b*SEQ + i
  const int b = rowi >> 10, i = rowi & (SEQ - 1);
  const int t = threadIdx.x;
  float4 v = {0.f, 0.f, 0.f, 0.f};
  if (i < CNT[b]) {
    int src = b * SEQ + IDX[rowi];
    v = *(const float4*)(ENC + (size_t)src * DMODEL + 4 * t);
  }
  *(float4*)(COMP + (size_t)rowi * DMODEL + 4 * t) = v;
}

// ---------------------------------------------------------------------------
// Dechunk: sequential EMA-style scan over s, parallel over (b, d)
// ---------------------------------------------------------------------------
__global__ __launch_bounds__(512) void k_dechunk(const float* __restrict__ MAINB,
                                                 const int* __restrict__ MK,
                                                 const int* __restrict__ CUM,
                                                 const float* __restrict__ BP,
                                                 float* __restrict__ OUT) {
  const int b = blockIdx.x, d = threadIdx.x;
  float prev = 0.f;
  for (int s = 0; s < SEQ; ++s) {
    int pos = b * SEQ + s;
    int ci = CUM[pos] - 1;
    if (ci < 0) ci = 0;
    float c = MAINB[(size_t)(b * SEQ + ci) * DMODEL + d];
    float p = BP[pos];
    float o = MK[pos] ? c : (p * c + (1.0f - p) * prev);
    OUT[(size_t)pos * DMODEL + d] = o;
    prev = o;
  }
}

// ---------------------------------------------------------------------------
// Host orchestration
// ---------------------------------------------------------------------------
extern "C" void kernel_launch(void* const* d_in, const int* in_sizes, int n_in,
                              void* d_out, int out_size, void* d_ws, size_t ws_size,
                              hipStream_t stream) {
  (void)out_size; (void)ws_size;
  char* ws = (char*)d_ws;
  size_t off = 0;
  auto alloc = [&](size_t bytes) -> void* {
    void* p = ws + off;
    off += (bytes + 255) & ~(size_t)255;
    return p;
  };

  // --- convert every weight tensor (inputs 1..n_in-1) to bf16 ---
  __bf16* wbf[64] = {};
  for (int i = 1; i < n_in; ++i) {
    int n = in_sizes[i];
    wbf[i] = (__bf16*)alloc((size_t)n * 2);
    k_f2bf<<<dim3((n / 4 + 255) / 256), dim3(256), 0, stream>>>(
        (const float*)d_in[i], wbf[i], n);
  }

  // --- scratch activations ---
  float*  xnf   = (float*)alloc((size_t)NTOK * DMODEL * 4);
  __bf16* xnb   = (__bf16*)alloc((size_t)NTOK * DMODEL * 2);
  float*  qbuf  = (float*)alloc((size_t)NTOK * DMODEL * 4);
  float*  kvbuf = (float*)alloc((size_t)NTOK * 2 * DMODEL * 4);
  __bf16* aoutb = (__bf16*)alloc((size_t)NTOK * DMODEL * 2);
  float*  hbuf  = (float*)alloc((size_t)NTOK * DMODEL * 4);
  float*  t1    = (float*)alloc((size_t)NTOK * DFFN * 4);
  float*  t3    = (float*)alloc((size_t)NTOK * DFFN * 4);
  __bf16* gb    = (__bf16*)alloc((size_t)NTOK * DFFN * 2);
  float*  we    = (float*)alloc((size_t)NTOK * NEXP * 4);
  float*  enc   = (float*)alloc((size_t)NTOK * DMODEL * 4);
  __bf16* encb  = (__bf16*)alloc((size_t)NTOK * DMODEL * 2);
  float*  qrb   = (float*)alloc((size_t)NTOK * DMODEL * 4);
  float*  krb   = (float*)alloc((size_t)NTOK * DMODEL * 4);
  float*  comp  = (float*)alloc((size_t)NTOK * DMODEL * 4);
  float*  maino = (float*)alloc((size_t)NTOK * DMODEL * 4);
  float*  decin = (float*)alloc((size_t)NTOK * DMODEL * 4);
  float*  bp    = (float*)alloc((size_t)NTOK * 4);
  int*    mk    = (int*)alloc((size_t)NTOK * 4);
  int*    cum   = (int*)alloc((size_t)NTOK * 4);
  int*    idx   = (int*)alloc((size_t)NTOK * 4);
  int*    cnt   = (int*)alloc(64);

  auto gemm = [&](int epi, const __bf16* A, const __bf16* Bw, float* C,
                  const float* R, const float* sc, int sstr, int N, int M, int K) {
    dim3 g(N / 128, M / 256), blk(256);
    if (epi == 0)      k_gemm<0><<<g, blk, 0, stream>>>(A, Bw, C, R, sc, sstr, N, M, K);
    else if (epi == 1) k_gemm<1><<<g, blk, 0, stream>>>(A, Bw, C, R, sc, sstr, N, M, K);
    else               k_gemm<2><<<g, blk, 0, stream>>>(A, Bw, C, R, sc, sstr, N, M, K);
  };

  // one transformer block: X (f32, NTOK x D) -> Y (f32, NTOK x D)
  auto run_block = [&](int bi, const float* X, float* Y) {
    // ---- MLA ----
    k_rmsnorm<<<NTOK, 128, 0, stream>>>(X, xnf, xnb);
    gemm(0, xnb, wbf[bi + 0], qbuf, nullptr, nullptr, 0, NTOK, DMODEL, DMODEL);      // q
    gemm(0, xnb, wbf[bi + 1], kvbuf, nullptr, nullptr, 0, NTOK, 2 * DMODEL, DMODEL); // kv
    k_attn<<<NTOK / 4, 128, 0, stream>>>(qbuf, kvbuf, aoutb);
    gemm(1, aoutb, wbf[bi + 2], hbuf, X, nullptr, 0, NTOK, DMODEL, DMODEL);          // wo + res
    // ---- MoE ----
    k_rmsnorm<<<NTOK, 128, 0, stream>>>(hbuf, xnf, xnb);
    k_gate<<<NTOK / 8, 256, 0, stream>>>(xnf, (const float*)d_in[bi + 3], we);
    hipMemcpyAsync(Y, hbuf, (size_t)NTOK * DMODEL * 4, hipMemcpyDeviceToDevice, stream);
    for (int e = 0; e < NEXP + 1; ++e) {  // 4 routed + shared
      int w1i = (e < NEXP) ? (bi + 4 + 3 * e) : (bi + 16);
      gemm(0, xnb, wbf[w1i + 0], t1, nullptr, nullptr, 0, NTOK, DFFN, DMODEL);
      gemm(0, xnb, wbf[w1i + 2], t3, nullptr, nullptr, 0, NTOK, DFFN, DMODEL);
      k_gelu_mul<<<(NTOK * DFFN / 4 + 255) / 256, 256, 0, stream>>>(t1, t3, gb, NTOK * DFFN);
      const float* sc = (e < NEXP) ? (we + e) : nullptr;
      gemm(2, gb, wbf[w1i + 1], Y, nullptr, sc, (e < NEXP) ? NEXP : 0, NTOK, DMODEL, DFFN);
    }
  };

  // d_in layout: 0=x; encoder@1, main@20, decoder@39 (19 tensors each); router wq@58, wk@59
  const float* x = (const float*)d_in[0];

  // encoder
  run_block(1, x, enc);

  // router (wq/wk projections then adjacent cosine)
  k_f2bf<<<(NTOK * DMODEL / 4 + 255) / 256, 256, 0, stream>>>(enc, encb, NTOK * DMODEL);
  gemm(0, encb, wbf[58], qrb, nullptr, nullptr, 0, NTOK, DMODEL, DMODEL);
  gemm(0, encb, wbf[59], krb, nullptr, nullptr, 0, NTOK, DMODEL, DMODEL);
  k_router<<<NTOK / 8, 256, 0, stream>>>(qrb, krb, bp, mk);
  k_scan_mask<<<1, NBATCH, 0, stream>>>(mk, cum, idx, cnt);
  k_chunk<<<NTOK, 128, 0, stream>>>(enc, idx, cnt, comp);

  // main block on zero-padded compressed sequence (graph-capture safe: no
  // host sync on data-dependent max_len; dechunk only reads valid rows)
  run_block(20, comp, maino);

  // dechunk
  k_dechunk<<<NBATCH, 512, 0, stream>>>(maino, mk, cum, bp, decin);

  // decoder -> final output
  run_block(39, decin, (float*)d_out);
}
